// SamplerGNN_48155173323055
// MI455X (gfx1250) — matrix operand order repaired and checked by reading, not compiled, not run on hardware
//
#include <hip/hip_runtime.h>
#include <hip/hip_bf16.h>

// ---------------------------------------------------------------------------
// GATv2 GNN for gfx1250 (MI455X).  wave32, WMMA f32_16x16x32_f16 for all
// dense GEMMs (node transforms + MLP head); atomic segment-softmax for the
// irregular edge aggregation (memory-bound gather/scatter, no matrix shape).
// ---------------------------------------------------------------------------

#define N_NODES 10000
#define N_EDGES 160000
#define ETOT    (N_EDGES + N_NODES)   // with self loops
#define BATCH   16
#define HC      16
#define HEADS   4
#define HID     128
#define NUM_OUT 30

typedef __attribute__((ext_vector_type(16))) _Float16 v16h;
typedef __attribute__((ext_vector_type(8)))  float    v8f;

__device__ __forceinline__ v8f wmma_f16(v16h a, v16h b, v8f c) {
  // D = A(16x32 f16) * B(32x16 f16) + C(16x16 f32)
  return __builtin_amdgcn_wmma_f32_16x16x32_f16(
      /*neg_a=*/false, a, /*neg_b=*/false, b,
      /*c_mod=*/(short)0, c, /*reuse_a=*/false, /*reuse_b=*/false);
}

// monotonic float<->u32 encoding so atomicMax(u32) == float max
__device__ __forceinline__ unsigned enc_f32(float f) {
  unsigned u = __float_as_uint(f);
  return (u & 0x80000000u) ? ~u : (u | 0x80000000u);
}
__device__ __forceinline__ float dec_f32(unsigned e) {
  unsigned u = (e & 0x80000000u) ? (e & 0x7fffffffu) : ~e;
  return __uint_as_float(u);
}

// ---------------------------- init kernels ---------------------------------
__global__ void __launch_bounds__(256) init_f32(float* p, float v, int n) {
  int i = blockIdx.x * 256 + threadIdx.x;
  if (i < n) p[i] = v;
}
__global__ void __launch_bounds__(256) init_u32(unsigned* p, unsigned v, int n) {
  int i = blockIdx.x * 256 + threadIdx.x;
  if (i < n) p[i] = v;
}
// h_next[b][n][c] = bias[c]   (aggregation accumulator, bias pre-added)
__global__ void __launch_bounds__(256) init_bias16(float* p, const float* __restrict__ bias, int n) {
  int i = blockIdx.x * 256 + threadIdx.x;
  if (i < n) p[i] = bias[i & 15];
}

// --------------------- node transform: xl/xr = x@W + b ---------------------
// One wave per 16-row tile; one v_wmma per (tile, W).  F is 32 (layer 0,
// exact K) or 16 (layers 1-3, zero-padded K).  Weights staged to LDS as f16.
__global__ void __launch_bounds__(256)
gat_transform(const float* __restrict__ in,
              const float* __restrict__ Wl, const float* __restrict__ Wr,
              const float* __restrict__ bl, const float* __restrict__ br,
              float* __restrict__ xl, float* __restrict__ xr,
              int F, int ntiles)
{
  __shared__ _Float16 sWl[32 * 16];
  __shared__ _Float16 sWr[32 * 16];
  __shared__ float    sbl[16], sbr[16];

  int tid = threadIdx.x;
  for (int i = tid; i < 32 * 16; i += 256) {
    int k = i >> 4, n = i & 15;
    sWl[i] = (k < F) ? (_Float16)Wl[k * 16 + n] : (_Float16)0.0f;
    sWr[i] = (k < F) ? (_Float16)Wr[k * 16 + n] : (_Float16)0.0f;
  }
  if (tid < 16) { sbl[tid] = bl[tid]; sbr[tid] = br[tid]; }
  __syncthreads();

  int wave = tid >> 5, lane = tid & 31;
  int tile = blockIdx.x * 8 + wave;
  if (tile >= ntiles) return;

  int M    = lane & 15;            // row within tile
  int base = (lane >> 4) * 8;      // K sub-block: lanes<16 -> K0..7/16..23, lanes>=16 -> K8..15/24..31
  const float* row = in + (size_t)(tile * 16 + M) * F;

  v16h a;
  if (F == 32) {
    #pragma unroll
    for (int j = 0; j < 8; ++j) {
      a[j]     = (_Float16)row[base + j];
      a[8 + j] = (_Float16)row[16 + base + j];
    }
  } else { // F == 16, pad K 16->32
    #pragma unroll
    for (int j = 0; j < 8; ++j) {
      a[j]     = (_Float16)row[base + j];
      a[8 + j] = (_Float16)0.0f;
    }
  }

  int n    = lane & 15;            // output column
  int koff = (lane >> 4) * 8;      // B: lanes<16 hold K0..7/16..23, lanes>=16 hold K8..15/24..31
  v16h bfl, bfr;
  #pragma unroll
  for (int t = 0; t < 16; ++t) {
    bfl[t] = sWl[(koff + t + (t & 8)) * 16 + n];
    bfr[t] = sWr[(koff + t + (t & 8)) * 16 + n];
  }

  v8f cl, cr;
  float vl = sbl[n], vr = sbr[n];
  #pragma unroll
  for (int r = 0; r < 8; ++r) { cl[r] = vl; cr[r] = vr; }

  cl = wmma_f16(a, bfl, cl);
  cr = wmma_f16(a, bfr, cr);

  int   Mb = (lane >> 4) * 8;      // D rows: lanes<16 -> M0..7, lanes>=16 -> M8..15
  float* ol = xl + (size_t)tile * 256;
  float* orr = xr + (size_t)tile * 256;
  #pragma unroll
  for (int r = 0; r < 8; ++r) {
    ol[(Mb + r) * 16 + n]  = cl[r];
    orr[(Mb + r) * 16 + n] = cr[r];
  }
}

// ------------------- edge pass A: scores + segment max ---------------------
__global__ void __launch_bounds__(256)
edge_score_max(const float* __restrict__ xl, const float* __restrict__ xr,
               const int* __restrict__ src, const int* __restrict__ dst,
               const float* __restrict__ att, float* __restrict__ ebuf,
               unsigned* __restrict__ mbits)
{
  int idx = blockIdx.x * 256 + threadIdx.x;
  if (idx >= ETOT * BATCH) return;
  int b = idx / ETOT, e = idx % ETOT;
  int s = (e < N_EDGES) ? src[e] : (e - N_EDGES);
  int d = (e < N_EDGES) ? dst[e] : (e - N_EDGES);

  const float4* pl = (const float4*)(xl + ((size_t)b * N_NODES + s) * HC);
  const float4* pr = (const float4*)(xr + ((size_t)b * N_NODES + d) * HC);
  unsigned* mb = mbits + ((size_t)b * N_NODES + d) * HEADS;

  float sc[4];
  #pragma unroll
  for (int h = 0; h < HEADS; ++h) {
    float4 l = pl[h], r = pr[h];
    float4 at = ((const float4*)att)[h];
    float z0 = l.x + r.x; z0 = z0 > 0.f ? z0 : 0.2f * z0;
    float z1 = l.y + r.y; z1 = z1 > 0.f ? z1 : 0.2f * z1;
    float z2 = l.z + r.z; z2 = z2 > 0.f ? z2 : 0.2f * z2;
    float z3 = l.w + r.w; z3 = z3 > 0.f ? z3 : 0.2f * z3;
    float sv = z0 * at.x + z1 * at.y + z2 * at.z + z3 * at.w;
    sc[h] = sv;
    atomicMax(&mb[h], enc_f32(sv));
  }
  ((float4*)ebuf)[idx] = make_float4(sc[0], sc[1], sc[2], sc[3]);
}

// ------------------- edge pass B: exp(e - max), segment sum ----------------
__global__ void __launch_bounds__(256)
edge_expsum(float* __restrict__ ebuf, const int* __restrict__ dst,
            const unsigned* __restrict__ mbits, float* __restrict__ ssum)
{
  int idx = blockIdx.x * 256 + threadIdx.x;
  if (idx >= ETOT * BATCH) return;
  int b = idx / ETOT, e = idx % ETOT;
  int d = (e < N_EDGES) ? dst[e] : (e - N_EDGES);

  float4 v = ((const float4*)ebuf)[idx];
  float sc[4] = { v.x, v.y, v.z, v.w };
  const unsigned* mb = mbits + ((size_t)b * N_NODES + d) * HEADS;
  float* sp = ssum + ((size_t)b * N_NODES + d) * HEADS;

  #pragma unroll
  for (int h = 0; h < HEADS; ++h) {
    float m  = dec_f32(mb[h]);
    float ex = __expf(sc[h] - m);
    sc[h] = ex;
    atomicAdd(&sp[h], ex);
  }
  ((float4*)ebuf)[idx] = make_float4(sc[0], sc[1], sc[2], sc[3]);
}

// ------------------- edge pass C: alpha-weighted scatter-add ---------------
__global__ void __launch_bounds__(256)
edge_aggregate(const float* __restrict__ ebuf, const float* __restrict__ ssum,
               const float* __restrict__ xl, const int* __restrict__ src,
               const int* __restrict__ dst, float* __restrict__ hout)
{
  int idx = blockIdx.x * 256 + threadIdx.x;
  if (idx >= ETOT * BATCH) return;
  int b = idx / ETOT, e = idx % ETOT;
  int s = (e < N_EDGES) ? src[e] : (e - N_EDGES);
  int d = (e < N_EDGES) ? dst[e] : (e - N_EDGES);

  float4 exv = ((const float4*)ebuf)[idx];
  float ex[4] = { exv.x, exv.y, exv.z, exv.w };
  const float* sp = ssum + ((size_t)b * N_NODES + d) * HEADS;
  const float4* pl = (const float4*)(xl + ((size_t)b * N_NODES + s) * HC);
  float* po = hout + ((size_t)b * N_NODES + d) * HC;

  #pragma unroll
  for (int h = 0; h < HEADS; ++h) {
    float alpha = ex[h] / (sp[h] + 1e-16f);
    float4 l = pl[h];
    atomicAdd(&po[h * 4 + 0], alpha * l.x);
    atomicAdd(&po[h * 4 + 1], alpha * l.y);
    atomicAdd(&po[h * 4 + 2], alpha * l.z);
    atomicAdd(&po[h * 4 + 3], alpha * l.w);
  }
}

// ------------------- mean pool (sum; divide by N in head) ------------------
__global__ void __launch_bounds__(256)
pool_sum(const float* __restrict__ h, float* __restrict__ pooled, int nblocks)
{
  int t = threadIdx.x;          // t = b*16 + c
  int b = t >> 4, c = t & 15;
  int per = (N_NODES + nblocks - 1) / nblocks;
  int n0 = blockIdx.x * per;
  int n1 = n0 + per; if (n1 > N_NODES) n1 = N_NODES;
  float acc = 0.f;
  for (int n = n0; n < n1; ++n)
    acc += h[((size_t)b * N_NODES + n) * HC + c];
  atomicAdd(&pooled[t], acc);
}

// ------------------- MLP head: feat -> 128 -> 128 -> 30 (WMMA) -------------
__global__ void __launch_bounds__(256)
head_mlp(const float* __restrict__ pooled, const float* __restrict__ obs,
         const float* __restrict__ Wp, const float* __restrict__ bp,
         const float* __restrict__ W1, const float* __restrict__ b1,
         const float* __restrict__ W2, const float* __restrict__ b2,
         const float* __restrict__ W3, const float* __restrict__ b3,
         float* __restrict__ out)
{
  __shared__ float cin[16 * 128];   // [feat15 | obs100 | pad] per batch row
  __shared__ float z1[16 * 128];
  __shared__ float z2[16 * 128];
  int tid = threadIdx.x;

  for (int i = tid; i < 16 * 128; i += 256) cin[i] = 0.f;
  __syncthreads();

  if (tid < 16 * 15) {                    // feat = mean @ Wp + bp
    int b = tid / 15, j = tid % 15;
    float acc = bp[j];
    #pragma unroll
    for (int c = 0; c < 16; ++c)
      acc += (pooled[b * 16 + c] * (1.0f / (float)N_NODES)) * Wp[c * 15 + j];
    cin[b * 128 + j] = acc;
  }
  for (int i = tid; i < 16 * 100; i += 256) {
    int b = i / 100, o = i % 100;
    cin[b * 128 + 15 + o] = obs[b * 100 + o];
  }
  __syncthreads();

  int wave  = tid >> 5, lane = tid & 31;
  int M     = lane & 15;
  int col   = wave * 16 + (lane & 15);
  int base  = (lane >> 4) * 8;
  int khalf = (lane >> 4) * 8;
  int Mb    = (lane >> 4) * 8;
  v8f acc;

  // GEMM1: [16,115(pad128)] x [115,128] + b1, tanh
  { float bv = b1[col];
    #pragma unroll
    for (int r = 0; r < 8; ++r) acc[r] = bv; }
  #pragma unroll
  for (int q = 0; q < 4; ++q) {
    v16h a, bb;
    #pragma unroll
    for (int j = 0; j < 8; ++j) {
      a[j]     = (_Float16)cin[M * 128 + q * 32 + base + j];
      a[8 + j] = (_Float16)cin[M * 128 + q * 32 + 16 + base + j];
    }
    #pragma unroll
    for (int t = 0; t < 16; ++t) {
      int k = q * 32 + khalf + t + (t & 8);
      bb[t] = (k < 115) ? (_Float16)W1[k * 128 + col] : (_Float16)0.0f;
    }
    acc = wmma_f16(a, bb, acc);
  }
  #pragma unroll
  for (int r = 0; r < 8; ++r) z1[(Mb + r) * 128 + col] = tanhf(acc[r]);
  __syncthreads();

  // GEMM2: [16,128] x [128,128] + b2, tanh
  { float bv = b2[col];
    #pragma unroll
    for (int r = 0; r < 8; ++r) acc[r] = bv; }
  #pragma unroll
  for (int q = 0; q < 4; ++q) {
    v16h a, bb;
    #pragma unroll
    for (int j = 0; j < 8; ++j) {
      a[j]     = (_Float16)z1[M * 128 + q * 32 + base + j];
      a[8 + j] = (_Float16)z1[M * 128 + q * 32 + 16 + base + j];
    }
    #pragma unroll
    for (int t = 0; t < 16; ++t)
      bb[t] = (_Float16)W2[(q * 32 + khalf + t + (t & 8)) * 128 + col];
    acc = wmma_f16(a, bb, acc);
  }
  #pragma unroll
  for (int r = 0; r < 8; ++r) z2[(Mb + r) * 128 + col] = tanhf(acc[r]);
  __syncthreads();

  // GEMM3: [16,128] x [128,30] + b3  (waves 0,1 cover 32 cols, mask >=30)
  if (wave < 2) {
    int c3 = wave * 16 + (lane & 15);
    float bv = (c3 < NUM_OUT) ? b3[c3] : 0.f;
    #pragma unroll
    for (int r = 0; r < 8; ++r) acc[r] = bv;
    #pragma unroll
    for (int q = 0; q < 4; ++q) {
      v16h a, bb;
      #pragma unroll
      for (int j = 0; j < 8; ++j) {
        a[j]     = (_Float16)z2[M * 128 + q * 32 + base + j];
        a[8 + j] = (_Float16)z2[M * 128 + q * 32 + 16 + base + j];
      }
      #pragma unroll
      for (int t = 0; t < 16; ++t) {
        int k = q * 32 + khalf + t + (t & 8);
        bb[t] = (c3 < NUM_OUT) ? (_Float16)W3[k * NUM_OUT + c3] : (_Float16)0.0f;
      }
      acc = wmma_f16(a, bb, acc);
    }
    if (c3 < NUM_OUT) {
      #pragma unroll
      for (int r = 0; r < 8; ++r) out[(Mb + r) * NUM_OUT + c3] = acc[r];
    }
  }
}

// ---------------------------------------------------------------------------
extern "C" void kernel_launch(void* const* d_in, const int* in_sizes, int n_in,
                              void* d_out, int out_size, void* d_ws, size_t ws_size,
                              hipStream_t stream) {
  const float* x   = (const float*)d_in[0];
  const float* obs = (const float*)d_in[1];
  const int*   src = (const int*)d_in[2];
  const int*   dst = (const int*)d_in[3];
  const float* W0l = (const float*)d_in[4];
  const float* W0r = (const float*)d_in[5];
  const float* b0l = (const float*)d_in[6];
  const float* b0r = (const float*)d_in[7];
  const float* Wl  = (const float*)d_in[8];
  const float* Wr  = (const float*)d_in[9];
  const float* bl  = (const float*)d_in[10];
  const float* br  = (const float*)d_in[11];
  const float* att = (const float*)d_in[12];
  const float* bias= (const float*)d_in[13];
  const float* Wp  = (const float*)d_in[14];
  const float* bp  = (const float*)d_in[15];
  const float* W1  = (const float*)d_in[16];
  const float* b1  = (const float*)d_in[17];
  const float* W2  = (const float*)d_in[18];
  const float* b2  = (const float*)d_in[19];
  const float* W3  = (const float*)d_in[20];
  const float* b3  = (const float*)d_in[21];

  // ---- workspace layout (~90 MB total) ----
  const size_t nNodeF = (size_t)BATCH * N_NODES * HC;     // 2,560,000
  const size_t nSeg   = (size_t)BATCH * N_NODES * HEADS;  //   640,000
  const int    EB     = ETOT * BATCH;                     // 2,720,000
  float*    h_a    = (float*)d_ws;
  float*    h_b    = h_a + nNodeF;
  float*    xl     = h_b + nNodeF;
  float*    xr     = xl + nNodeF;
  float*    ssum   = xr + nNodeF;
  unsigned* mbits  = (unsigned*)(ssum + nSeg);
  float*    ebuf   = (float*)(mbits + nSeg);
  float*    pooled = ebuf + (size_t)EB * HEADS;

  const int gSeg  = (int)((nSeg + 255) / 256);
  const int gNode = (int)((nNodeF + 255) / 256);
  const int gEB   = (EB + 255) / 256;

  float* hcur = h_a;
  float* hnext = h_b;
  for (int l = 0; l < 4; ++l) {
    const float* in  = (l == 0) ? x : hcur;
    const int    F   = (l == 0) ? 32 : 16;
    const float* wlp = (l == 0) ? W0l : (Wl + (size_t)(l - 1) * HC * HC);
    const float* wrp = (l == 0) ? W0r : (Wr + (size_t)(l - 1) * HC * HC);
    const float* blp = (l == 0) ? b0l : (bl + (size_t)(l - 1) * HC);
    const float* brp = (l == 0) ? b0r : (br + (size_t)(l - 1) * HC);

    gat_transform<<<1250, 256, 0, stream>>>(in, wlp, wrp, blp, brp, xl, xr, F, 10000);
    init_u32<<<gSeg, 256, 0, stream>>>(mbits, 0u, (int)nSeg);
    init_f32<<<gSeg, 256, 0, stream>>>(ssum, 0.f, (int)nSeg);
    init_bias16<<<gNode, 256, 0, stream>>>(hnext, bias + (size_t)l * HC, (int)nNodeF);
    edge_score_max<<<gEB, 256, 0, stream>>>(xl, xr, src, dst, att + (size_t)l * HC, ebuf, mbits);
    edge_expsum<<<gEB, 256, 0, stream>>>(ebuf, dst, mbits, ssum);
    edge_aggregate<<<gEB, 256, 0, stream>>>(ebuf, ssum, xl, src, dst, hnext);

    float* t = hcur; hcur = hnext; hnext = t;
  }

  init_f32<<<1, 256, 0, stream>>>(pooled, 0.f, BATCH * HC);
  pool_sum<<<64, 256, 0, stream>>>(hcur, pooled, 64);
  head_mlp<<<1, 256, 0, stream>>>(pooled, obs, Wp, bp, W1, b1, W2, b2, W3, b3,
                                  (float*)d_out);
}
